// EnergyGuidedRouter_15075335209576
// MI455X (gfx1250) — matrix operand order repaired and checked by reading, not compiled, not run on hardware
//
#include <hip/hip_runtime.h>
#include <hip/hip_bf16.h>

// ============================================================================
// EnergyGuidedRouter for MI455X (gfx1250, wave32, WMMA + TDM).
//
// Algebraic optimization: (rw @ ao) @ w_p == rw @ (ao @ w_p), shrinking the
// dominant 68.7 GFLOP GEMM to 4.3 GFLOP; the pipeline is then HBM-bound
// (~0.4 GB traffic -> ~17 us at 23.3 TB/s).  All matmuls use
// v_wmma_f32_16x16x32_f16 (f16 operands, f32 accumulate); every WMMA operand
// except attention's V is fetched with 128-bit loads (weights pre-transposed to [N,K]).
//
// The anchor-aggregation kernel (the big x-streaming GEMM) stages its x panel
// into LDS with the Tensor Data Mover (tensor_load_to_lds + s_wait_tensorcnt),
// replacing per-lane VMEM staging with a single DMA descriptor per panel.
// This toolchain exposes the 6-argument builtin (extra int32x8 group, zeroed).
// ============================================================================

typedef __attribute__((ext_vector_type(16))) _Float16 v16h;
typedef __attribute__((ext_vector_type(8)))  _Float16 v8h;
typedef __attribute__((ext_vector_type(8)))  float    v8f;
typedef __attribute__((ext_vector_type(4)))  float    v4f;
typedef __attribute__((ext_vector_type(4)))  unsigned int u32x4;
typedef __attribute__((ext_vector_type(8)))  int      i32x8;
typedef __attribute__((ext_vector_type(4)))  int      i32x4;

#define DEV __device__ __forceinline__

#if __has_builtin(__builtin_amdgcn_tensor_load_to_lds) && \
    __has_builtin(__builtin_amdgcn_s_wait_tensorcnt)
#define USE_TDM 1
#else
#define USE_TDM 0
#endif

constexpr int BB = 8, SS = 4096, DD = 1024, KK = 64, NH = 8, HD = 128;
constexpr int TK = 128;   // k-rows of x staged per TDM/coop panel in anchor GEMM

// ---------------------------------------------------------------------------
// WMMA 16x16x32 f16 fragments (wave32), contiguous k order used here (A and B share it):
//  A (16x32 MxK): lane L: m=L&15, kbA=(L<16)?0:16, elem i: k = kbA + i
//    -> 16 contiguous halves at +kbA (two b128 loads).
//  B (32x16 KxN): lane L: n=L&15, kbB=(L<16)?0:16, elem i: k = kbB + i
//    -> 16 contiguous halves when B is stored [N,K] (two b128 loads).
//  C/D (16x16 f32): lane L: n=L&15; VGPR v holds m = v + ((L<16)?0:8).
// ---------------------------------------------------------------------------
DEV v8f wmma32(const v16h& a, const v16h& b, const v8f& c) {
  return __builtin_amdgcn_wmma_f32_16x16x32_f16(
      /*neg_a=*/false, a, /*neg_b=*/false, b,
      /*c_mod=*/(short)0, c, /*reuse_a=*/false, /*reuse_b=*/false);
}

DEV v16h frag_a_f16(const _Float16* p) {            // 16 contiguous halves
  v8h lo = *reinterpret_cast<const v8h*>(p);
  v8h hi = *reinterpret_cast<const v8h*>(p + 8);
  v16h r;
#pragma unroll
  for (int i = 0; i < 8; ++i) { r[i] = lo[i]; r[i + 8] = hi[i]; }
  return r;
}

DEV v16h frag_a_f32(const float* p) {               // x operand: f32 -> f16
  v4f a0 = *reinterpret_cast<const v4f*>(p);
  v4f a1 = *reinterpret_cast<const v4f*>(p + 4);
  v4f b0 = *reinterpret_cast<const v4f*>(p + 8);
  v4f b1 = *reinterpret_cast<const v4f*>(p + 12);
  v16h r;
#pragma unroll
  for (int i = 0; i < 4; ++i) {
    r[i]      = (_Float16)a0[i];
    r[i + 4]  = (_Float16)a1[i];
    r[i + 8]  = (_Float16)b0[i];
    r[i + 12] = (_Float16)b1[i];
  }
  return r;
}

DEV v16h frag_b16(const _Float16* p) {              // 16 contiguous halves
  v8h lo = *reinterpret_cast<const v8h*>(p);
  v8h hi = *reinterpret_cast<const v8h*>(p + 8);
  v16h r;
#pragma unroll
  for (int i = 0; i < 8; ++i) { r[i] = lo[i]; r[i + 8] = hi[i]; }
  return r;
}

template <typename CF>
DEV void wmma_store_tile(const v8f& acc, CF cf) {
  const int lane = threadIdx.x & 31;
  const int n    = lane & 15;
  const int mb   = (lane < 16) ? 0 : 8;
#pragma unroll
  for (int v = 0; v < 8; ++v) cf(mb + v, n, acc[v]);
}

// ---------------------------------------------------------------------------
// Kernel 1: weights f32 -> f16, transposed to [N,K] for contiguous B frags.
// ---------------------------------------------------------------------------
__global__ void prep_f16(const float* w1, const float* w2, const float* wqkv,
                         const float* wo, const float* wp,
                         _Float16* w1t, _Float16* w2t, _Float16* wqkvt,
                         _Float16* wot, _Float16* wpt) {
  const size_t i      = (size_t)blockIdx.x * blockDim.x + threadIdx.x;
  const size_t stride = (size_t)gridDim.x * blockDim.x;
  for (size_t j = i; j < (size_t)KK * DD; j += stride) {
    size_t n = j / DD, k = j % DD;
    w1t[j] = (_Float16)w1[k * KK + n];
  }
  for (size_t j = i; j < (size_t)KK * KK; j += stride) {
    size_t n = j / KK, k = j % KK;
    w2t[j] = (_Float16)w2[k * KK + n];
  }
  for (size_t j = i; j < (size_t)3 * DD * DD; j += stride) {
    size_t n = j / DD, k = j % DD;
    wqkvt[j] = (_Float16)wqkv[k * (3 * DD) + n];
  }
  for (size_t j = i; j < (size_t)DD * DD; j += stride) {
    size_t n = j / DD, k = j % DD;
    wot[j] = (_Float16)wo[k * DD + n];
  }
  for (size_t j = i; j < (size_t)DD * DD; j += stride) {
    size_t n = j / DD, k = j % DD;
    wpt[j] = (_Float16)wp[k * DD + n];
  }
}

// ---------------------------------------------------------------------------
// Kernel 2: routing. block = 128 (4 waves), 16 tokens per block.
// ---------------------------------------------------------------------------
__global__ void routing_kernel(const float* __restrict__ x,
                               const float* __restrict__ efas,
                               const _Float16* __restrict__ w1t,
                               const float* __restrict__ b1,
                               const _Float16* __restrict__ w2t,
                               const float* __restrict__ b2,
                               const float* __restrict__ w_e,
                               const float* __restrict__ b_e,
                               _Float16* __restrict__ rwh,
                               _Float16* __restrict__ rwt) {
  __shared__ __align__(16) _Float16 hs[16][KK];
  __shared__ float lg[16][KK];
  const int t0   = blockIdx.x * 16;
  const int wave = threadIdx.x >> 5;
  const int lane = threadIdx.x & 31;
  const int mn   = lane & 15;
  const int kbA  = (lane < 16) ? 0 : 16;
  const int kbB  = (lane < 16) ? 0 : 16;
  const int n0   = wave * 16;

  const float*    arow = x + ((size_t)t0 + mn) * DD;
  const _Float16* brow = w1t + (size_t)(n0 + mn) * DD;
  v8f acc = {};
  for (int k0 = 0; k0 < DD; k0 += 32)
    acc = wmma32(frag_a_f32(arow + k0 + kbA), frag_b16(brow + k0 + kbB), acc);
  wmma_store_tile(acc, [&](int m, int n, float v) {
    float h = v + b1[n0 + n];
    hs[m][n0 + n] = (_Float16)fmaxf(h, 0.f);
  });
  __syncthreads();

  const _Float16* b2row = w2t + (size_t)(n0 + mn) * KK;
  v8f acc2 = {};
  for (int k0 = 0; k0 < KK; k0 += 32)
    acc2 = wmma32(frag_a_f16(&hs[mn][k0 + kbA]), frag_b16(b2row + k0 + kbB), acc2);
  wmma_store_tile(acc2, [&](int m, int n, float v) {
    const int nn = n0 + n;
    const float e = efas[t0 + m] * w_e[nn] + b_e[nn];
    lg[m][nn] = 2.f * e + v + b2[nn];
  });
  __syncthreads();

  if (threadIdx.x < 16) {
    const int m  = threadIdx.x;
    const int bi = t0 / SS;
    const int s  = (t0 % SS) + m;
    float mx = -1e30f;
    for (int n = 0; n < KK; ++n) mx = fmaxf(mx, lg[m][n]);
    float sum = 0.f;
    for (int n = 0; n < KK; ++n) sum += __expf((lg[m][n] - mx) * 10.f);
    const float inv = 1.f / sum;
    for (int n = 0; n < KK; ++n) {
      const _Float16 rv = (_Float16)(__expf((lg[m][n] - mx) * 10.f) * inv);
      rwh[(size_t)(t0 + m) * KK + n]      = rv;   // [token][anchor]
      rwt[((size_t)bi * KK + n) * SS + s] = rv;   // [b][anchor][s]
    }
  }
}

// ---------------------------------------------------------------------------
// Kernel 3: anchors a[b] = rw[b]^T (64 x S) @ x[b] (S x D).
// A from rwt (contiguous b128); B = x panel [TK x 16] staged in LDS.
// TDM path: one tensor_load_to_lds per panel (2-D descriptor, 4B elements,
// tile 16 x TK, tensor_dim0_stride = 1024), wave 0 issues, TENSORcnt wait.
// grid = (D/16, B); block = 128 (4 waves share the panel).
// ---------------------------------------------------------------------------
__global__ void anchor_kernel(const float* __restrict__ x,
                              const _Float16* __restrict__ rwt,
                              _Float16* __restrict__ ah) {
#if USE_TDM
  __shared__ __align__(16) float    xt[TK][16];   // raw f32 panel from TDM (8 KB)
#endif
  __shared__ __align__(16) _Float16 bt[16][TK];   // f16 panel, [n][k] (4 KB)
  const int b    = blockIdx.y;
  const int nt   = blockIdx.x;
  const int wave = threadIdx.x >> 5;
  const int lane = threadIdx.x & 31;
  const int mn   = lane & 15;
  const int kbA  = (lane < 16) ? 0 : 16;
  const int kbB  = (lane < 16) ? 0 : 16;
  const int m0   = wave * 16;

  const float*    xb   = x + (size_t)b * SS * DD + nt * 16;
  const _Float16* arow = rwt + ((size_t)b * KK + m0 + mn) * SS;

#if USE_TDM
  // LDS byte offset of the destination tile (AS(3) pointers are LDS offsets).
  const unsigned lds_off =
      (unsigned)(unsigned long long)(__attribute__((address_space(3))) void*)&xt[0][0];
#endif

  v8f acc = {};
  for (int kt = 0; kt < SS; kt += TK) {
    __syncthreads();   // previous panel fully consumed
#if USE_TDM
    if (wave == 0) {
      const unsigned long long ga = (unsigned long long)(xb + (size_t)kt * DD);
      // D# group 0: count=1 | lds_addr | global_addr | type=2 (ISA 8.3)
      u32x4 g0 = { 1u, lds_off, (unsigned)ga,
                   (unsigned)((ga >> 32) & 0x1FFFFFFu) | (2u << 30) };
      // D# group 1: data_size=4B; tensor 16 x TK; tile 16 x TK; stride0=1024
      i32x8 g1 = { (int)(2u << 16),           // data_size = 2 (4 bytes)
                   (int)(16u << 16),          // tensor_dim0 = 16  (bits 63:48)
                   (int)((unsigned)TK << 16), // tensor_dim1 = TK  (bits 95:80)
                   (int)(16u << 16),          // tile_dim0 = 16    (bits 127:112)
                   (int)TK,                   // tile_dim1 = TK    (bits 143:128)
                   (int)DD,                   // tensor_dim0_stride (bits 191:160)
                   0, 0 };
      i32x4 gz  = { 0, 0, 0, 0 };             // groups 2/3 unused (2-D tile)
      i32x8 gz8 = { 0, 0, 0, 0, 0, 0, 0, 0 }; // extra group (clang-23 6-arg form)
      __builtin_amdgcn_tensor_load_to_lds(g0, g1, gz, gz, gz8, 0);
      __builtin_amdgcn_s_wait_tensorcnt(0);
    }
    __syncthreads();   // panel visible to all waves
    // convert f32 panel -> f16 [n][k] so B fragments stay 128-bit LDS loads
    for (int t = threadIdx.x; t < (TK * 16) / 4; t += 128) {
      const int kk = t >> 2, nq = (t & 3) * 4;
      v4f v = *reinterpret_cast<const v4f*>(&xt[kk][nq]);
#pragma unroll
      for (int j = 0; j < 4; ++j) bt[nq + j][kk] = (_Float16)v[j];
    }
#else
    // cooperative staging fallback: float4 global loads, f16 transpose in LDS
    for (int t = threadIdx.x; t < (TK * 16) / 4; t += 128) {
      const int kk = t >> 2, nq = (t & 3) * 4;
      v4f v = *reinterpret_cast<const v4f*>(&xb[(size_t)(kt + kk) * DD + nq]);
#pragma unroll
      for (int j = 0; j < 4; ++j) bt[nq + j][kk] = (_Float16)v[j];
      if (kt + TK < SS)
        __builtin_prefetch(&xb[(size_t)(kt + TK + kk) * DD + nq], 0, 0);
    }
#endif
    __syncthreads();   // f16 panel ready
#pragma unroll
    for (int k0 = 0; k0 < TK; k0 += 32)
      acc = wmma32(frag_a_f16(arow + kt + k0 + kbA),
                   frag_b16(&bt[mn][k0 + kbB]), acc);
  }
  wmma_store_tile(acc, [&](int m, int n, float v) {
    ah[((size_t)b * KK + m0 + m) * DD + nt * 16 + n] = (_Float16)v;
  });
}

// ---------------------------------------------------------------------------
// Kernel 4 (shared): Out[b] = A[b] (64 x Kd) @ Wt^T (+bias).  Wt is [N,K].
// transOut: store Out as [b][N][64] (for the final GEMM's B operand).
// ---------------------------------------------------------------------------
__global__ void gemm64_kernel(const _Float16* __restrict__ A,
                              const _Float16* __restrict__ Wt,
                              const float* __restrict__ bias,
                              _Float16* __restrict__ Out,
                              int Kd, int N, int transOut) {
  const int b    = blockIdx.y;
  const int nt   = blockIdx.x;
  const int wave = threadIdx.x >> 5;
  const int lane = threadIdx.x & 31;
  const int mn   = lane & 15;
  const int kbA  = (lane < 16) ? 0 : 16;
  const int kbB  = (lane < 16) ? 0 : 16;
  const int m0   = wave * 16;

  const _Float16* arow = A + ((size_t)b * KK + m0 + mn) * Kd;
  const _Float16* brow = Wt + (size_t)(nt * 16 + mn) * Kd;
  v8f acc = {};
  for (int k0 = 0; k0 < Kd; k0 += 32)
    acc = wmma32(frag_a_f16(arow + k0 + kbA), frag_b16(brow + k0 + kbB), acc);
  wmma_store_tile(acc, [&](int m, int n, float v) {
    const int nn = nt * 16 + n;
    const float o = v + (bias ? bias[nn] : 0.f);
    if (transOut) Out[((size_t)b * N + nn) * KK + m0 + m] = (_Float16)o;
    else          Out[((size_t)b * KK + m0 + m) * N + nn] = (_Float16)o;
  });
}

// ---------------------------------------------------------------------------
// Kernel 5: attention per (batch, head).  block = 256 (8 waves).
// ---------------------------------------------------------------------------
__global__ void attn_kernel(const _Float16* __restrict__ qkvh,
                            _Float16* __restrict__ hoh) {
  __shared__ float sc[KK][KK];
  __shared__ __align__(16) _Float16 pr[KK][KK];
  const int b    = blockIdx.x >> 3;
  const int h    = blockIdx.x & 7;
  const int wave = threadIdx.x >> 5;
  const int lane = threadIdx.x & 31;
  const int mn   = lane & 15;
  const int kbA  = (lane < 16) ? 0 : 16;
  const int kbB  = (lane < 16) ? 0 : 16;
  const _Float16* qk = qkvh + (size_t)b * KK * (3 * DD);

  for (int tid = wave; tid < 16; tid += 8) {
    const int mt = tid >> 2, ct = tid & 3;
    const _Float16* qrow = qk + (size_t)(mt * 16 + mn) * 3072 + h * HD;
    const _Float16* krow = qk + (size_t)(ct * 16 + mn) * 3072 + DD + h * HD;
    v8f acc = {};
    for (int k0 = 0; k0 < HD; k0 += 32)
      acc = wmma32(frag_a_f16(qrow + k0 + kbA), frag_b16(krow + k0 + kbB), acc);
    wmma_store_tile(acc, [&](int m, int n, float v) {
      sc[mt * 16 + m][ct * 16 + n] = v * 0.08838834764831845f;  // 1/sqrt(128)
    });
  }
  __syncthreads();

  if (threadIdx.x < KK) {
    const int m = threadIdx.x;
    float mx = -1e30f;
    for (int n = 0; n < KK; ++n) mx = fmaxf(mx, sc[m][n]);
    float sum = 0.f;
    for (int n = 0; n < KK; ++n) sum += __expf(sc[m][n] - mx);
    const float inv = 1.f / sum;
    for (int n = 0; n < KK; ++n) pr[m][n] = (_Float16)(__expf(sc[m][n] - mx) * inv);
  }
  __syncthreads();

  for (int tid = wave; tid < 32; tid += 8) {
    const int mt = tid >> 3, ct = tid & 7;
    v8f acc = {};
    for (int k0 = 0; k0 < KK; k0 += 32) {
      v16h bf;
#pragma unroll
      for (int i = 0; i < 16; ++i)
        bf[i] = qk[(size_t)(k0 + kbB + i) * 3072 + 2 * DD + h * HD + ct * 16 + mn];
      acc = wmma32(frag_a_f16(&pr[mt * 16 + mn][k0 + kbA]), bf, acc);
    }
    wmma_store_tile(acc, [&](int m, int n, float v) {
      hoh[((size_t)b * KK + mt * 16 + m) * DD + h * HD + ct * 16 + n] = (_Float16)v;
    });
  }
}

// ---------------------------------------------------------------------------
// Kernel 6: out[b] = rw[b] (S x 64) @ aop[b] (64 x D) + b_p.
// ---------------------------------------------------------------------------
__global__ void final_kernel(const _Float16* __restrict__ rwh,
                             const _Float16* __restrict__ aopt,
                             const float* __restrict__ b_p,
                             float* __restrict__ out) {
  const int b    = blockIdx.z;
  const int nt   = blockIdx.x;
  const int wave = threadIdx.x >> 5;
  const int lane = threadIdx.x & 31;
  const int mn   = lane & 15;
  const int kbA  = (lane < 16) ? 0 : 16;
  const int kbB  = (lane < 16) ? 0 : 16;
  const int s0   = blockIdx.y * 64 + wave * 16;

  const _Float16* arow = rwh + ((size_t)b * SS + s0 + mn) * KK;
  const _Float16* brow = aopt + ((size_t)b * DD + nt * 16 + mn) * KK;
  v8f acc = {};
  for (int k0 = 0; k0 < KK; k0 += 32)
    acc = wmma32(frag_a_f16(arow + k0 + kbA), frag_b16(brow + k0 + kbB), acc);
  wmma_store_tile(acc, [&](int m, int n, float v) {
    const int nn = nt * 16 + n;
    out[((size_t)b * SS + s0 + m) * DD + nn] = v + b_p[nn];
  });
}

// ---------------------------------------------------------------------------
// Host-side launch.
// ---------------------------------------------------------------------------
extern "C" void kernel_launch(void* const* d_in, const int* in_sizes, int n_in,
                              void* d_out, int out_size, void* d_ws, size_t ws_size,
                              hipStream_t stream) {
  const float* x      = (const float*)d_in[0];
  const float* efas   = (const float*)d_in[1];
  const float* w_e    = (const float*)d_in[2];
  const float* b_e    = (const float*)d_in[3];
  const float* w1     = (const float*)d_in[4];
  const float* b1     = (const float*)d_in[5];
  const float* w2     = (const float*)d_in[6];
  const float* b2     = (const float*)d_in[7];
  const float* w_qkv  = (const float*)d_in[8];
  const float* b_qkv  = (const float*)d_in[9];
  const float* w_o    = (const float*)d_in[10];
  const float* b_o    = (const float*)d_in[11];
  const float* w_p    = (const float*)d_in[12];
  const float* b_p    = (const float*)d_in[13];
  float* out = (float*)d_out;

  _Float16* w = (_Float16*)d_ws;
  _Float16* w1t   = w; w += (size_t)KK * DD;
  _Float16* w2t   = w; w += (size_t)KK * KK;
  _Float16* wqkvt = w; w += (size_t)3 * DD * DD;
  _Float16* wot   = w; w += (size_t)DD * DD;
  _Float16* wpt   = w; w += (size_t)DD * DD;
  _Float16* rwh   = w; w += (size_t)BB * SS * KK;
  _Float16* rwt   = w; w += (size_t)BB * KK * SS;
  _Float16* ah    = w; w += (size_t)BB * KK * DD;
  _Float16* qkvh  = w; w += (size_t)BB * KK * 3 * DD;
  _Float16* hoh   = w; w += (size_t)BB * KK * DD;
  _Float16* aoh   = w; w += (size_t)BB * KK * DD;
  _Float16* aopt  = w; w += (size_t)BB * DD * KK;

  prep_f16<<<2048, 256, 0, stream>>>(w1, w2, w_qkv, w_o, w_p,
                                     w1t, w2t, wqkvt, wot, wpt);
  routing_kernel<<<(BB * SS) / 16, 128, 0, stream>>>(
      x, efas, w1t, b1, w2t, b2, w_e, b_e, rwh, rwt);
  anchor_kernel<<<dim3(DD / 16, BB), 128, 0, stream>>>(x, rwt, ah);
  gemm64_kernel<<<dim3(3 * DD / 16, BB), 128, 0, stream>>>(
      ah, wqkvt, b_qkv, qkvh, DD, 3 * DD, 0);
  attn_kernel<<<BB * NH, 256, 0, stream>>>(qkvh, hoh);
  gemm64_kernel<<<dim3(DD / 16, BB), 128, 0, stream>>>(
      hoh, wot, b_o, aoh, DD, DD, 0);
  gemm64_kernel<<<dim3(DD / 16, BB), 128, 0, stream>>>(
      aoh, wpt, nullptr, aopt, DD, DD, 1);
  final_kernel<<<dim3(DD / 16, SS / 64, BB), 128, 0, stream>>>(
      rwh, aopt, b_p, out);

  (void)in_sizes; (void)n_in; (void)out_size; (void)ws_size;
}
